// FeatureAlign_CA_c2f_Block5_1932735283803
// MI455X (gfx1250) — matrix operand order repaired and checked by reading, not compiled, not run on hardware
//
#include <hip/hip_runtime.h>
#include <math.h>

typedef __attribute__((ext_vector_type(16))) _Float16 v16h;
typedef __attribute__((ext_vector_type(8)))  _Float16 v8h;
typedef __attribute__((ext_vector_type(8)))  float    v8f;
typedef __attribute__((ext_vector_type(4)))  unsigned int v4u;
typedef __attribute__((ext_vector_type(8)))  int      v8i;
typedef __attribute__((ext_vector_type(4)))  int      v4i;

#define BATCH 2

// ---------------------------------------------------------------------------
// f32 -> f16 weight conversion
// ---------------------------------------------------------------------------
__global__ void k_cvt_f16(const float* __restrict__ src, _Float16* __restrict__ dst,
                          int n)
{
  int t = blockIdx.x * blockDim.x + threadIdx.x;
  if (t < n) dst[t] = (_Float16)src[t];
}

// ---------------------------------------------------------------------------
// Generic conv (1x1 or 3x3, pad=ksize/2) as WMMA GEMM, f16 weights.
// Block = 128 threads (4 waves). One block: 64 out-channels x 16 positions.
//   stage 1: cooperative im2col -> LDS f16 B panel (chunked layout so each
//            lane's 16 B-halves are two contiguous b128 LDS loads)
//   stage 2: per-wave v_wmma_f32_16x16x32_f16 over K
// act: 0=none, 1=relu, 2=prelu(alpha[0])
// ---------------------------------------------------------------------------
__global__ void __launch_bounds__(128)
k_conv_wmma(const float* __restrict__ x, const _Float16* __restrict__ wh,
            const float* __restrict__ bias, float* __restrict__ out,
            int Cin, int Cout, int H, int W, int ksize, int act,
            const float* __restrict__ alpha)
{
  __shared__ __align__(16) _Float16 smemB[27648];   // up to K=1728 x 16

  const int HW  = H * W;
  const int ntN = (HW + 15) >> 4;
  const int b   = blockIdx.x / ntN;
  const int tn  = blockIdx.x - b * ntN;
  const int ksq = ksize * ksize;
  const int K   = Cin * ksq;
  const float* xb = x + (size_t)b * Cin * HW;

  // ---- stage 1: im2col into LDS ----
  for (int idx = threadIdx.x; idx < K * 16; idx += 128) {
    int n = idx & 15;
    int s = idx >> 4;                    // k_lin
    int p = (tn << 4) + n;
    float f = 0.0f;
    if (p < HW) {
      int py0 = p / W, px0 = p - py0 * W;
      int ci = s / ksq, r = s - ci * ksq;
      int yy = py0, xx = px0;
      if (ksize == 3) { int ry = r / 3; yy += ry - 1; xx += (r - ry * 3) - 1; }
      if (yy >= 0 && yy < H && xx >= 0 && xx < W)
        f = xb[(size_t)ci * HW + yy * W + xx];
    }
    smemB[((s >> 5) << 9) + (n << 5) + (s & 31)] = (_Float16)f;
  }
  __syncthreads();

  // ---- stage 2: WMMA GEMM ----
  const int wave = threadIdx.x >> 5;
  const int lane = threadIdx.x & 31;
  const int mrow = lane & 15;
  const int kb8  = (lane >> 4) << 3;
  const int ncol = lane & 15;
  const int kb16 = (lane >> 4) << 4;
  const int o16  = ((blockIdx.y << 2) + wave) << 4;
  const int orow = o16 + mrow;
  const int orc  = (orow < Cout) ? orow : (Cout - 1);  // clamp; rows dropped at store

  v8f acc = {};
  for (int kc = 0; kc < K; kc += 32) {
    const _Float16* ap = wh + (size_t)orc * K + kc + kb16;
    __builtin_prefetch(ap + 64, 0, 0);                 // global_prefetch_b8
    v8h a0 = *(const v8h*)ap;                          // k = kc+kb16 .. +7
    v8h a1 = *(const v8h*)(ap + 8);                    // k = kc+kb16+8 .. +15
    const _Float16* bp = smemB + ((kc >> 5) << 9) + (ncol << 5) + kb16;
    v8h b0 = *(const v8h*)bp;                          // k = kc+kb16 .. +7
    v8h b1 = *(const v8h*)(bp + 8);                    // k = kc+kb16+8 .. +15
    v16h av = __builtin_shufflevector(a0, a1, 0,1,2,3,4,5,6,7,8,9,10,11,12,13,14,15);
    v16h bv = __builtin_shufflevector(b0, b1, 0,1,2,3,4,5,6,7,8,9,10,11,12,13,14,15);
    acc = __builtin_amdgcn_wmma_f32_16x16x32_f16(false, av, false, bv,
                                                 (short)0, acc, false, false);
  }

  const int p = (tn << 4) + ncol;
#pragma unroll
  for (int r = 0; r < 8; ++r) {
    int o = o16 + r + ((lane >> 4) << 3);
    if (o < Cout && p < HW) {
      float v = acc[r] + bias[o];
      if (act == 1)      v = fmaxf(v, 0.0f);
      else if (act == 2) { float a = alpha[0]; v = (v >= 0.0f) ? v : a * v; }
      out[((size_t)b * Cout + o) * HW + p] = v;
    }
  }
}

// ---------------------------------------------------------------------------
// Modulated deformable conv (C=64, G=8, 3x3) fused with ReLU.
//   TDM: wave0 DMAs the full 64x576 f16 weight panel into LDS (overlapped
//        with stage 1), tracked by TENSORcnt.
//   stage 1: 128 threads build masked bilinear samples -> LDS f16 B panel
//   stage 2: 4 waves x (16x16xK=576) WMMA, A and B both from LDS
// ---------------------------------------------------------------------------
__global__ void __launch_bounds__(128)
k_dcn_wmma(const float* __restrict__ x, const float* __restrict__ om,
           const _Float16* __restrict__ wh, const float* __restrict__ bias,
           float* __restrict__ out, int H, int W)
{
  __shared__ __align__(16) _Float16 smemB[9216];     // 576 x 16 panel (chunked)
  __shared__ __align__(16) _Float16 smemW[36864];    // 64 x 576 weights

  const int HW  = H * W;
  const int ntN = (HW + 15) >> 4;
  const int b   = blockIdx.x / ntN;
  const int tn  = blockIdx.x - b * ntN;

#if __has_builtin(__builtin_amdgcn_tensor_load_to_lds)
  if (threadIdx.x < 32) {   // wave 0 issues the TDM weight fetch
    unsigned int lds = (unsigned int)(size_t)(&smemW[0]) & 0xFFFFFu;
    unsigned long long ga = (unsigned long long)(size_t)wh;
    v4u g0;
    g0[0] = 1u;                                        // count=1, user D#
    g0[1] = lds;                                       // lds_addr
    g0[2] = (unsigned int)ga;                          // global_addr lo
    g0[3] = ((unsigned int)(ga >> 32) & 0x01FFFFFFu) | (2u << 30); // hi | type=2
    v8i g1;
    g1[0] = (int)0x00010000u;          // data_size = 2 bytes
    g1[1] = (int)(36864u << 16);       // tensor_dim0 lo16 (=0x9000)
    g1[2] = (int)(1u << 16);           // tensor_dim0 hi16=0, tensor_dim1 lo16=1
    g1[3] = (int)(0x9000u << 16);      // tensor_dim1 hi16=0, tile_dim0=36864
    g1[4] = 1;                         // tile_dim1=1, tile_dim2=0
    g1[5] = 36864;                     // tensor_dim0_stride lo32
    g1[6] = 0;
    g1[7] = 0;
    v4i gz; gz[0] = 0; gz[1] = 0; gz[2] = 0; gz[3] = 0;
#if defined(__clang_major__) && (__clang_major__ >= 23)
    v8i g4; g4[0]=0; g4[1]=0; g4[2]=0; g4[3]=0; g4[4]=0; g4[5]=0; g4[6]=0; g4[7]=0;
    __builtin_amdgcn_tensor_load_to_lds(g0, g1, gz, gz, g4, 0);
#else
    __builtin_amdgcn_tensor_load_to_lds(g0, g1, gz, gz, 0);
#endif
  }
#else
  for (int idx = threadIdx.x; idx < 36864 / 8; idx += 128)
    ((v8h*)smemW)[idx] = ((const v8h*)wh)[idx];
#endif

  const float* omb = om + (size_t)b * 216 * HW;
  const float* xb  = x  + (size_t)b * 64  * HW;

  // ---- stage 1: masked bilinear samples -> LDS B panel ----
  for (int idx = threadIdx.x; idx < 576 * 16; idx += 128) {
    int n = idx & 15;
    int s = idx >> 4;                 // k_lin = ci*9 + kk
    int p = (tn << 4) + n;
    float val = 0.0f;
    if (p < HW) {
      int yy = p / W, xx = p - (p / W) * W;
      int ci = s / 9, kk = s - ci * 9;
      int g  = ci >> 3;
      int gk = g * 9 + kk;
      float dy = omb[(size_t)gk * HW + p];
      float dx = omb[(size_t)(72 + gk) * HW + p];
      float mk = 1.0f / (1.0f + __expf(-omb[(size_t)(144 + gk) * HW + p]));
      int ry = kk / 3;
      float py = (float)yy + (float)(ry - 1) + dy;
      float px = (float)xx + (float)((kk - ry * 3) - 1) + dx;
      float fy0 = floorf(py), fx0 = floorf(px);
      float wy = py - fy0, wx = px - fx0;
      const float* xc = xb + (size_t)ci * HW;
      float a = 0.0f;
#pragma unroll
      for (int cy = 0; cy < 2; ++cy) {
        float yv  = fy0 + (float)cy;
        float wyy = cy ? wy : (1.0f - wy);
#pragma unroll
        for (int cx = 0; cx < 2; ++cx) {
          float xv  = fx0 + (float)cx;
          float wxx = cx ? wx : (1.0f - wx);
          float valid = (yv >= 0.0f && yv <= (float)(H - 1) &&
                         xv >= 0.0f && xv <= (float)(W - 1)) ? 1.0f : 0.0f;
          float ycl = fminf(fmaxf(yv, 0.0f), (float)(H - 1));
          float xcl = fminf(fmaxf(xv, 0.0f), (float)(W - 1));
          int yi = (int)ycl, xi = (int)xcl;
          a += xc[yi * W + xi] * (wyy * wxx * valid);
        }
      }
      val = a * mk;
    }
    smemB[((s >> 5) << 9) + (n << 5) + (s & 31)] = (_Float16)val;
  }

#if __has_builtin(__builtin_amdgcn_tensor_load_to_lds)
  if (threadIdx.x < 32) __builtin_amdgcn_s_wait_tensorcnt(0);
#endif
  __syncthreads();

  // ---- stage 2: WMMA, A and B from LDS ----
  const int wave = threadIdx.x >> 5;
  const int lane = threadIdx.x & 31;
  const int mrow = lane & 15;
  const int kb8  = (lane >> 4) << 3;
  const int ncol = lane & 15;
  const int kb16 = (lane >> 4) << 4;
  const int o16  = wave << 4;

  v8f acc = {};
  for (int kc = 0; kc < 576; kc += 32) {
    const _Float16* ap = smemW + (size_t)(o16 + mrow) * 576 + kc + kb16;
    v8h a0 = *(const v8h*)ap;
    v8h a1 = *(const v8h*)(ap + 8);
    const _Float16* bp = smemB + ((kc >> 5) << 9) + (ncol << 5) + kb16;
    v8h b0 = *(const v8h*)bp;
    v8h b1 = *(const v8h*)(bp + 8);
    v16h av = __builtin_shufflevector(a0, a1, 0,1,2,3,4,5,6,7,8,9,10,11,12,13,14,15);
    v16h bv = __builtin_shufflevector(b0, b1, 0,1,2,3,4,5,6,7,8,9,10,11,12,13,14,15);
    acc = __builtin_amdgcn_wmma_f32_16x16x32_f16(false, av, false, bv,
                                                 (short)0, acc, false, false);
  }

  const int p = (tn << 4) + ncol;
#pragma unroll
  for (int r = 0; r < 8; ++r) {
    int o = o16 + r + ((lane >> 4) << 3);
    if (p < HW)
      out[((size_t)b * 64 + o) * HW + p] = fmaxf(acc[r] + bias[o], 0.0f);
  }
}

// ---------------------------------------------------------------------------
// Elementwise / glue kernels
// ---------------------------------------------------------------------------
__global__ void k_concat(float* __restrict__ dst, const float* __restrict__ rgb,
                         const float* __restrict__ dep, const float* __restrict__ offup,
                         int catC, int HW)
{
  int total = BATCH * 64 * HW;
  int t = blockIdx.x * blockDim.x + threadIdx.x;
  if (t >= total) return;
  int p = t % HW; int tmp = t / HW; int c = tmp % 64; int b = tmp / 64;
  size_t base = (size_t)b * catC * HW;
  dst[base + (size_t)c * HW + p]        = rgb[t];
  dst[base + (size_t)(64 + c) * HW + p] = 2.0f * dep[t];
  if (catC == 192) dst[base + (size_t)(128 + c) * HW + p] = offup[t];
}

__global__ void k_resize(const float* __restrict__ src, float* __restrict__ dst,
                         int C, int H, int W, int Hn, int Wn)
{
  int total = BATCH * C * Hn * Wn;
  int t = blockIdx.x * blockDim.x + threadIdx.x;
  if (t >= total) return;
  int xn = t % Wn; int tmp = t / Wn; int yn = tmp % Hn; tmp /= Hn;
  int c = tmp % C; int b = tmp / C;
  float sy = (Hn > 1) ? (float)(H - 1) / (float)(Hn - 1) : 0.0f;
  float sx = (Wn > 1) ? (float)(W - 1) / (float)(Wn - 1) : 0.0f;
  float ys = yn * sy, xs = xn * sx;
  int y0 = (int)floorf(ys); int y1 = (y0 + 1 < H) ? y0 + 1 : H - 1; float wy = ys - y0;
  int x0 = (int)floorf(xs); int x1 = (x0 + 1 < W) ? x0 + 1 : W - 1; float wx = xs - x0;
  const float* s = src + ((size_t)b * C + c) * H * W;
  float t0 = s[y0 * W + x0] * (1.0f - wy) + s[y1 * W + x0] * wy;
  float t1 = s[y0 * W + x1] * (1.0f - wy) + s[y1 * W + x1] * wy;
  dst[((size_t)b * C + c) * Hn * Wn + yn * Wn + xn] = t0 * (1.0f - wx) + t1 * wx;
}

__global__ void k_maxpool8(const float* __restrict__ src, float* __restrict__ dst,
                           int H, int W)
{
  int h = H >> 3, w = W >> 3;
  int total = BATCH * 64 * h * w;
  int t = blockIdx.x * blockDim.x + threadIdx.x;
  if (t >= total) return;
  int xo = t % w; int tmp = t / w; int yo = tmp % h; tmp /= h;
  int c = tmp % 64; int b = tmp / 64;
  const float* s = src + ((size_t)b * 64 + c) * H * W;
  float m = -3.4e38f;
  for (int dy = 0; dy < 8; ++dy)
    for (int dx = 0; dx < 8; ++dx)
      m = fmaxf(m, s[(yo * 8 + dy) * W + xo * 8 + dx]);
  dst[((size_t)b * 64 + c) * h * w + yo * w + xo] = m;
}

// S[b,i,j] = sum_c Kt[b,c,i] * Q[b,c,j]
__global__ void k_qk(const float* __restrict__ Kt, const float* __restrict__ Q,
                     float* __restrict__ S, int n)
{
  int total = BATCH * n * n;
  int t = blockIdx.x * blockDim.x + threadIdx.x;
  if (t >= total) return;
  int j = t % n; int tmp = t / n; int i = tmp % n; int b = tmp / n;
  float s = 0.0f;
  for (int c = 0; c < 64; ++c)
    s += Kt[((size_t)b * 64 + c) * n + i] * Q[((size_t)b * 64 + c) * n + j];
  S[(size_t)b * n * n + (size_t)i * n + j] = s;
}

// softmax over i (axis=1) for each (b, j)
__global__ void k_softmax_col(float* __restrict__ S, int n)
{
  int total = BATCH * n;
  int t = blockIdx.x * blockDim.x + threadIdx.x;
  if (t >= total) return;
  int j = t % n; int b = t / n;
  float* col = S + (size_t)b * n * n + j;
  float mx = -3.4e38f;
  for (int i = 0; i < n; ++i) mx = fmaxf(mx, col[(size_t)i * n]);
  float sum = 0.0f;
  for (int i = 0; i < n; ++i) sum += __expf(col[(size_t)i * n] - mx);
  float inv = 1.0f / sum;
  for (int i = 0; i < n; ++i) col[(size_t)i * n] = __expf(col[(size_t)i * n] - mx) * inv;
}

// att[b,c,i] = sum_j P[b,i,j] * V[b,c,j]
__global__ void k_av(const float* __restrict__ P, const float* __restrict__ V,
                     float* __restrict__ att, int n)
{
  int total = BATCH * 64 * n;
  int t = blockIdx.x * blockDim.x + threadIdx.x;
  if (t >= total) return;
  int i = t % n; int tmp = t / n; int c = tmp % 64; int b = tmp / 64;
  const float* Pb = P + (size_t)b * n * n + (size_t)i * n;
  const float* Vb = V + ((size_t)b * 64 + c) * n;
  float s = 0.0f;
  for (int j = 0; j < n; ++j) s += Pb[j] * Vb[j];
  att[((size_t)b * 64 + c) * n + i] = s;
}

__global__ void k_fusein(const float* __restrict__ dep, const float* __restrict__ attd,
                         const float* __restrict__ col, const float* __restrict__ attc,
                         float* __restrict__ dst, int HW)
{
  int total = BATCH * 64 * HW;
  int t = blockIdx.x * blockDim.x + threadIdx.x;
  if (t >= total) return;
  int p = t % HW; int tmp = t / HW; int c = tmp % 64; int b = tmp / 64;
  size_t si = ((size_t)b * 64 + c) * HW + p;
  size_t base = (size_t)b * 128 * HW;
  dst[base + (size_t)c * HW + p]        = dep[si] + 1.5f * attd[si];
  dst[base + (size_t)(64 + c) * HW + p] = col[si] + 1.5f * attc[si];
}

// ---------------------------------------------------------------------------
// Orchestration
// ---------------------------------------------------------------------------
extern "C" void kernel_launch(void* const* d_in, const int* in_sizes, int n_in,
                              void* d_out, int out_size, void* d_ws, size_t ws_size,
                              hipStream_t stream)
{
  (void)in_sizes; (void)n_in; (void)out_size; (void)ws_size;

  const int    hwArr[5]  = {8, 16, 32, 64, 128};
  const size_t outOff[5] = {0, 8192, 40960, 172032, 696320};

  float* ws = (float*)d_ws;
  float* R0 = ws;                       // cat (B x 192 x 16384) / fuse_in
  float* R1 = R0 + 6291456;             // off
  float* R2 = R1 + 2097152;             // off_up
  float* R3 = R2 + 2097152;             // om (B x 216 x 16384) / attention scratch
  float* R4 = R3 + 7077888;             // fD
  _Float16* WH = (_Float16*)(R4 + 2097152);   // f16 weight pool

  // attention scratch (aliased into R3; om is dead by fusion time)
  float* attd_rs = R3;
  float* attc_rs = R3 + 2097152;
  float* sm      = R3 + 4194304;
  float* d_pool = sm;            float* c_pool = sm + 32768;
  float* dQ = sm +  65536;  float* dK = sm +  98304;  float* dV = sm + 131072;
  float* cQ = sm + 163840;  float* cK = sm + 196608;  float* cV = sm + 229376;
  float* attd = sm + 262144; float* attc = sm + 294912;
  float* Sbuf = sm + 327680; // B*256*256

  // ---- one-time (per launch) f16 weight conversion ----
  size_t woff = 0;
  auto cvt = [&](int idx, int nelem) -> _Float16* {
    _Float16* p = WH + woff; woff += (size_t)nelem;
    k_cvt_f16<<<(nelem + 255) / 256, 256, 0, stream>>>((const float*)d_in[idx], p, nelem);
    return p;
  };
  _Float16* ogwF[4]; _Float16* dwoF[3]; _Float16* dwF[3];
  _Float16* qkvF[4][6]; _Float16* fuF[4];
  ogwF[0] = cvt(10, 64 * 128);
  ogwF[1] = cvt(12, 64 * 192);
  ogwF[2] = cvt(14, 64 * 192);
  ogwF[3] = cvt(16, 64 * 192);
  for (int i = 0; i < 3; ++i) { dwoF[i] = cvt(18 + 4 * i, 216 * 576); dwF[i] = cvt(20 + 4 * i, 64 * 576); }
  for (int f = 0; f < 4; ++f) {
    for (int t = 0; t < 6; ++t) qkvF[f][t] = cvt(30 + 21 * f + 3 * t, 64 * 576);
    fuF[f] = cvt(30 + 21 * f + 18, 64 * 1152);
  }

  float* outp = (float*)d_out;

  for (int i = 0; i < 5; ++i) {
    const int H = hwArr[i], W = H, HW = H * W;
    const float* rgb = (const float*)d_in[2 * i];
    const float* dep = (const float*)d_in[2 * i + 1];
    const int catC = (i == 0) ? 128 : 192;
    const int ntN  = (HW + 15) / 16;

    // concat(rgb, 2*depth [, off_up])
    {
      int total = BATCH * 64 * HW;
      k_concat<<<(total + 255) / 256, 256, 0, stream>>>(
          R0, rgb, dep, (i == 0) ? (const float*)nullptr : R2, catC, HW);
    }

    // offset 1x1 conv -> R1
    const int ogSel = (i < 3) ? i : 3;
    const int ogB   = (i == 0) ? 11 : (i == 1) ? 13 : (i == 2) ? 15 : 17;
    k_conv_wmma<<<dim3(BATCH * ntN, 1), 128, 0, stream>>>(
        R0, ogwF[ogSel], (const float*)d_in[ogB], R1,
        catC, 64, H, W, 1, 0, nullptr);

    // DCN offset-map conv (64 -> 216, 3x3) -> R3
    const int dSel  = (i <= 2) ? i : 2;
    const int dbase = 18 + 4 * dSel;
    k_conv_wmma<<<dim3(BATCH * ntN, 4), 128, 0, stream>>>(
        R1, dwoF[dSel], (const float*)d_in[dbase + 1], R3,
        64, 216, H, W, 3, 0, nullptr);

    // deformable conv + relu -> R4 (TDM weight staging inside)
    k_dcn_wmma<<<BATCH * ntN, 128, 0, stream>>>(
        dep, R3, dwF[dSel], (const float*)d_in[dbase + 3], R4, H, W);

    // off_up for next level -> R2
    if (i < 4) {
      int Hn = hwArr[i + 1];
      int total = BATCH * 64 * Hn * Hn;
      k_resize<<<(total + 255) / 256, 256, 0, stream>>>(R1, R2, 64, H, W, Hn, Hn);
    }

    // -------- fusion (level 4 reuses f3 params) --------
    const int fSel = (i < 3) ? i : 3;
    const int fb   = 30 + 21 * fSel;
    const int h = H / 8, w2 = W / 8, n = h * w2;

    {
      int total = BATCH * 64 * n;
      int g = (total + 255) / 256;
      k_maxpool8<<<g, 256, 0, stream>>>(R4, d_pool, H, W);
      k_maxpool8<<<g, 256, 0, stream>>>(rgb, c_pool, H, W);
    }

    float* qkv[6] = {dQ, dK, dV, cQ, cK, cV};
    const int ntNs = (n + 15) / 16;
    for (int t = 0; t < 6; ++t) {
      const float* src = (t < 3) ? d_pool : c_pool;
      k_conv_wmma<<<dim3(BATCH * ntNs, 1), 128, 0, stream>>>(
          src, qkvF[fSel][t], (const float*)d_in[fb + 3 * t + 1],
          qkv[t], 64, 64, h, w2, 3, 2, (const float*)d_in[fb + 3 * t + 2]);
    }

    {
      int gQK = (BATCH * n * n + 255) / 256;
      int gSM = (BATCH * n + 255) / 256;
      int gAV = (BATCH * 64 * n + 255) / 256;
      int gRS = (BATCH * 64 * HW + 255) / 256;

      // mut_d = softmax(cK^T dQ); att_c = (mut_d @ cV^T)^T
      k_qk<<<gQK, 256, 0, stream>>>(cK, dQ, Sbuf, n);
      k_softmax_col<<<gSM, 256, 0, stream>>>(Sbuf, n);
      k_av<<<gAV, 256, 0, stream>>>(Sbuf, cV, attc, n);
      k_resize<<<gRS, 256, 0, stream>>>(attc, attc_rs, 64, h, w2, H, W);

      // mut_c = softmax(dK^T cQ); att_d = (mut_c @ dV^T)^T
      k_qk<<<gQK, 256, 0, stream>>>(dK, cQ, Sbuf, n);
      k_softmax_col<<<gSM, 256, 0, stream>>>(Sbuf, n);
      k_av<<<gAV, 256, 0, stream>>>(Sbuf, dV, attd, n);
      k_resize<<<gRS, 256, 0, stream>>>(attd, attd_rs, 64, h, w2, H, W);
    }

    // fuse_in = concat(fD + 1.5*att_d, rgb + 1.5*att_c) -> R0
    {
      int total = BATCH * 64 * HW;
      k_fusein<<<(total + 255) / 256, 256, 0, stream>>>(
          R4, attd_rs, rgb, attc_rs, R0, HW);
    }

    // final fuse conv (128 -> 64, 3x3, PReLU) straight into d_out
    k_conv_wmma<<<dim3(BATCH * ntN, 1), 128, 0, stream>>>(
        R0, fuF[fSel], (const float*)d_in[fb + 19],
        outp + outOff[i], 128, 64, H, W, 3, 2, (const float*)d_in[fb + 20]);
  }
}
